// ConvolutionalSelfAttention_3899830305058
// MI455X (gfx1250) — matrix operand order repaired and checked by reading, not compiled, not run on hardware
//
#include <hip/hip_runtime.h>
#include <hip/hip_bf16.h>
#include <math.h>

// ---------------------------------------------------------------------------
// ConvolutionalSelfAttention, algebraically reduced:
//   k = x@Wk.T+bk ; q = x@Wq.T+bq ; v = x@Wv.T+bv   (per position)
//   S[b] = normalize(k) @ normalize(q).T            (1024x1024 per batch)
//   wmap[b,m] = sum_n v[b,n]*softmax_n(S[b,:,m])    (column softmax, scalar V)
//   out[b,c,i,j] = sum_{3x3 window} wmap[b,n]*x[b,n,c]
// ---------------------------------------------------------------------------

#define CDIM 256
#define HWDIM 1024
#define BDIM 16
#define CHW 30

typedef __attribute__((ext_vector_type(16))) _Float16 v16h;
typedef __attribute__((ext_vector_type(8)))  _Float16 v8h;
typedef __attribute__((ext_vector_type(8)))  float    v8f;

union Frag16 { v16h v; v8h h[2]; };

// Build a 16x32 / 32x16 f16 WMMA fragment from two 8-half (16B) runs.
__device__ __forceinline__ v16h load_frag(const _Float16* p0, const _Float16* p1) {
    Frag16 f;
    f.h[0] = *(const v8h*)p0;
    f.h[1] = *(const v8h*)p1;
    return f.v;
}

// Async 16B copy global -> LDS (CDNA5 GLOBAL_LOAD_ASYNC_TO_LDS_B128, ASYNCcnt).
// lds_off is the wave-relative LDS byte offset; for generic pointers into
// __shared__ this is the low 32 bits.
__device__ __forceinline__ void async_copy16(unsigned int lds_off, const void* gptr) {
    unsigned long long ga = (unsigned long long)(uintptr_t)gptr;
    asm volatile("global_load_async_to_lds_b128 %0, %1, off"
                 :: "v"(lds_off), "v"(ga) : "memory");
}

__device__ __forceinline__ void wait_async0() {
    asm volatile("s_wait_asynccnt 0x0" ::: "memory");
}

// ---------------------------------------------------------------------------
// Stage 0: convert a weight matrix f32 -> f16
// ---------------------------------------------------------------------------
__global__ __launch_bounds__(256) void csa_cvt_w(const float* __restrict__ w,
                                                 _Float16* __restrict__ wh, int n) {
    int i = blockIdx.x * 256 + threadIdx.x;
    if (i < n) wh[i] = (_Float16)w[i];
}

// ---------------------------------------------------------------------------
// Stage 1: transpose batch [B,C,H*W] f32 -> xh [B,HW,C] f16 (tiled via LDS)
// ---------------------------------------------------------------------------
__global__ __launch_bounds__(256) void csa_transpose(const float* __restrict__ batch,
                                                     _Float16* __restrict__ xh) {
    __shared__ float tile[32][33];
    const int tx = threadIdx.x & 31;
    const int ty = threadIdx.x >> 5;
    const int b  = blockIdx.z;
    const int n0 = blockIdx.x * 32;
    const int c0 = blockIdx.y * 32;

    const float* src = batch + ((size_t)b * CDIM + c0) * HWDIM + n0;
    for (int r = 0; r < 4; ++r) {
        int cc = ty + r * 8;
        tile[cc][tx] = src[(size_t)cc * HWDIM + tx];   // coalesced over n
    }
    __syncthreads();
    _Float16* dst = xh + ((size_t)b * HWDIM + n0) * CDIM + c0;
    for (int r = 0; r < 4; ++r) {
        int nn = ty + r * 8;
        dst[(size_t)nn * CDIM + tx] = (_Float16)tile[tx][nn];  // coalesced over c
    }
}

// ---------------------------------------------------------------------------
// Stage 2: values v[b,n] = batch[b,:,n] . value_w + value_b  (exact f32)
// ---------------------------------------------------------------------------
__global__ __launch_bounds__(256) void csa_values(const float* __restrict__ batch,
                                                  const float* __restrict__ vw,
                                                  const float* __restrict__ vb,
                                                  float* __restrict__ vout) {
    __shared__ float w[CDIM];
    const int tid = threadIdx.x;
    w[tid] = vw[tid];
    __syncthreads();
    const int b = blockIdx.y;
    const int n = blockIdx.x * 256 + tid;
    const float* p = batch + (size_t)b * CDIM * HWDIM + n;
    float acc = 0.0f;
    for (int c = 0; c < CDIM; ++c) {
        if ((c & 31) == 0 && c + 32 < CDIM)
            __builtin_prefetch(p + (size_t)(c + 32) * HWDIM, 0, 1);  // global_prefetch_b8
        acc += p[(size_t)c * HWDIM] * w[c];   // coalesced over n
    }
    vout[b * HWDIM + n] = acc + vb[0];
}

// ---------------------------------------------------------------------------
// Stage 3: projection GEMM + bias + row L2 normalize -> f16 khat / qhat
// M-block = 32 rows of (b*HW+n), full N=256, K=256 via 8x wmma 16x16x32.
// 8 waves: wave w -> M-tile (w&1), 4 N-tiles starting at (w>>1)*4.
// ---------------------------------------------------------------------------
__global__ __launch_bounds__(256) void csa_proj_norm(const _Float16* __restrict__ xh,
                                                     const _Float16* __restrict__ wk,
                                                     const _Float16* __restrict__ wq,
                                                     const float* __restrict__ bk,
                                                     const float* __restrict__ bq,
                                                     _Float16* __restrict__ khat,
                                                     _Float16* __restrict__ qhat) {
    __shared__ __align__(16) _Float16 lds_a[32 * CDIM];   // 16 KB A panel
    __shared__ float lds_o[32 * 257];                     // f32 results (+pad)
    __shared__ float lds_s[32];                           // 1/norm per row

    const int tid  = threadIdx.x;
    const int wave = tid >> 5;
    const int lane = tid & 31;
    const int lo   = lane & 15;
    const int hi   = lane >> 4;

    const bool isQ = (blockIdx.y != 0);
    const _Float16* W    = isQ ? wq : wk;
    const float*    bias = isQ ? bq : bk;
    _Float16*       dst  = isQ ? qhat : khat;

    const int mbase = blockIdx.x * 32;     // row base in [0, B*HW)

    {   // async A-panel stage: 32x256 halves = 1024 x 16B, ASYNCcnt tracked
        const unsigned int abase = (unsigned int)(uintptr_t)lds_a;
        const char* src = (const char*)(xh + (size_t)mbase * CDIM);
        for (int i = tid; i < 32 * CDIM / 8; i += 256)
            async_copy16(abase + i * 16, src + (size_t)i * 16);
        wait_async0();
    }
    __syncthreads();

    const int mi  = wave & 1;
    const int nt0 = (wave >> 1) * 4;

    v8f zero = {};
    v8f acc[4];
    for (int t = 0; t < 4; ++t) acc[t] = zero;

    for (int s = 0; s < 8; ++s) {
        const int kb = s * 32;
        const _Float16* arow = lds_a + (mi * 16 + lo) * CDIM + kb + hi * 16;
        v16h afrag = load_frag(arow, arow + 8);                      // ds_load_b128 x2
        for (int t = 0; t < 4; ++t) {
            // B[k=i, j=o] = W[o][i]; lane col o = tile*16+lo, 16 contiguous halves
            const _Float16* brow = W + (size_t)((nt0 + t) * 16 + lo) * CDIM + kb + hi * 16;
            v16h bfrag = load_frag(brow, brow + 8);                  // global_load_b128 x2
            acc[t] = __builtin_amdgcn_wmma_f32_16x16x32_f16(
                false, afrag, false, bfrag, (short)0, acc[t], false, false);
        }
    }

    // epilogue: bias add, spill f32 tiles to LDS (D layout: row = r + hi*8)
    for (int t = 0; t < 4; ++t) {
        const int col = (nt0 + t) * 16 + lo;
        const float bv = bias[col];
        for (int r = 0; r < 8; ++r) {
            const int row = mi * 16 + r + hi * 8;
            lds_o[row * 257 + col] = acc[t][r] + bv;
        }
    }
    __syncthreads();

    if (tid < 32) {     // L2 norm per row; stride-257 rows are bank-conflict free
        float ss = 0.0f;
        const float* rp = lds_o + tid * 257;
        for (int i = 0; i < CDIM; ++i) { float v = rp[i]; ss += v * v; }
        lds_s[tid] = 1.0f / fmaxf(sqrtf(ss), 1e-12f);
    }
    __syncthreads();

    for (int i = tid; i < 32 * CDIM; i += 256) {
        const int row = i >> 8, col = i & 255;
        dst[(size_t)(mbase + row) * CDIM + col] =
            (_Float16)(lds_o[row * 257 + col] * lds_s[row]);
    }
}

// ---------------------------------------------------------------------------
// Stage 4: scalar-V attention (flash-style column softmax of khat@qhat^T)
// Per wave: 16 output columns m, resident q-fragments (64 VGPRs). Key panels
// (16 rows x 256) are double-buffered in LDS via async global->LDS copies
// overlapping the 8 WMMAs + online softmax of the current panel.
// ---------------------------------------------------------------------------
__global__ __launch_bounds__(256) void csa_attention(const _Float16* __restrict__ khat,
                                                     const _Float16* __restrict__ qhat,
                                                     const float* __restrict__ vvals,
                                                     float* __restrict__ wmap) {
    __shared__ __align__(16) _Float16 lds_k[2][16 * CDIM];  // 2 x 8 KB key panels
    __shared__ float lds_v[HWDIM];                          // 4 KB values

    const int tid  = threadIdx.x;
    const int wave = tid >> 5;
    const int lane = tid & 31;
    const int lo   = lane & 15;
    const int hi   = lane >> 4;
    const int b    = blockIdx.y;
    const int mw   = blockIdx.x * 128 + wave * 16;        // this wave's 16 columns

    for (int i = tid; i < HWDIM; i += 256) lds_v[i] = vvals[b * HWDIM + i];

    // resident B fragments: qhat row (mw+lo), one 32B run per K-step
    v16h bfrag[8];
    {
        const _Float16* qrow = qhat + (size_t)(b * HWDIM + mw + lo) * CDIM;
        for (int s = 0; s < 8; ++s) {
            const _Float16* p = qrow + s * 32 + hi * 16;
            bfrag[s] = load_frag(p, p + 8);
        }
    }

    const unsigned int kbase = (unsigned int)(uintptr_t)(&lds_k[0][0]);
    const char* ksrc = (const char*)(khat + (size_t)b * HWDIM * CDIM);

    // prime buffer 0 with panel nb=0
    for (int i = tid; i < 512; i += 256)
        async_copy16(kbase + i * 16, ksrc + (size_t)i * 16);
    wait_async0();
    __syncthreads();

    float runmax = -1.0e30f, runZ = 0.0f, runN = 0.0f;
    v8f zero = {};

    for (int nb = 0; nb < 64; ++nb) {
        const int cur = nb & 1;

        // kick off async fill of the other buffer with the next panel
        if (nb + 1 < 64) {
            const unsigned int nbase = kbase + (unsigned int)(1 - cur) * (16 * CDIM * 2);
            const char* nsrc = ksrc + (size_t)(nb + 1) * 16 * CDIM * 2;
            for (int i = tid; i < 512; i += 256)
                async_copy16(nbase + i * 16, nsrc + (size_t)i * 16);
        }

        // 8x WMMA on the current panel
        const _Float16* panel = &lds_k[cur][0];
        v8f acc = zero;
        for (int s = 0; s < 8; ++s) {
            const _Float16* arow = panel + lo * CDIM + s * 32 + hi * 16;
            v16h afrag = load_frag(arow, arow + 8);
            acc = __builtin_amdgcn_wmma_f32_16x16x32_f16(
                false, afrag, false, bfrag[s], (short)0, acc, false, false);
        }

        // online softmax over rows n = nb*16 + r + hi*8 for column (mw+lo)
        float cmax = runmax;
        for (int r = 0; r < 8; ++r) cmax = fmaxf(cmax, acc[r]);
        const float rescale = __expf(runmax - cmax);
        runZ *= rescale;
        runN *= rescale;
        const float* vp = lds_v + nb * 16 + hi * 8;        // LDS broadcast reads
        for (int r = 0; r < 8; ++r) {
            const float e = __expf(acc[r] - cmax);         // v_exp_f32
            runZ += e;
            runN += vp[r] * e;
        }
        runmax = cmax;

        wait_async0();       // next panel resident in LDS
        __syncthreads();     // all waves done with cur, next panel visible
    }

    // merge lane pairs (l, l^16) which cover the same column (wave32 shuffle)
    const float omax = __shfl_xor(runmax, 16, 32);
    const float oZ   = __shfl_xor(runZ, 16, 32);
    const float oN   = __shfl_xor(runN, 16, 32);
    const float m = fmaxf(runmax, omax);
    const float Z = runZ * __expf(runmax - m) + oZ * __expf(omax - m);
    const float N = runN * __expf(runmax - m) + oN * __expf(omax - m);
    if (hi == 0) wmap[b * HWDIM + mw + lo] = N / Z;
}

// ---------------------------------------------------------------------------
// Stage 5: out[b,c,i,j] = sum over 3x3 window of wmap[b,n]*batch[b,c,n]
// ---------------------------------------------------------------------------
__global__ __launch_bounds__(256) void csa_finalize(const float* __restrict__ batch,
                                                    const float* __restrict__ wmap,
                                                    float* __restrict__ out) {
    __shared__ float wimg[HWDIM];
    const int tid = threadIdx.x;
    const int b = blockIdx.y, c = blockIdx.x;
    const float* img = batch + ((size_t)b * CDIM + c) * HWDIM;
    const float* wm  = wmap + b * HWDIM;
    for (int i = tid; i < HWDIM; i += 256) wimg[i] = wm[i] * img[i];
    __syncthreads();
    float* o = out + ((size_t)b * CDIM + c) * (CHW * CHW);
    for (int oi = tid; oi < CHW * CHW; oi += 256) {
        const int i = oi / CHW, j = oi % CHW;
        float s = 0.0f;
        for (int di = 0; di < 3; ++di)
            for (int dj = 0; dj < 3; ++dj)
                s += wimg[(i + di) * 32 + (j + dj)];
        o[oi] = s;
    }
}

// ---------------------------------------------------------------------------
extern "C" void kernel_launch(void* const* d_in, const int* in_sizes, int n_in,
                              void* d_out, int out_size, void* d_ws, size_t ws_size,
                              hipStream_t stream) {
    const float* batch   = (const float*)d_in[0];   // [16,256,32,32]
    const float* key_w   = (const float*)d_in[1];   // [256,256]
    const float* key_b   = (const float*)d_in[2];   // [256]
    const float* query_w = (const float*)d_in[3];   // [256,256]
    const float* query_b = (const float*)d_in[4];   // [256]
    const float* value_w = (const float*)d_in[5];   // [256]
    const float* value_b = (const float*)d_in[6];   // [1]
    float* out = (float*)d_out;                     // [16,256,30,30]

    // workspace carve-up (~24.4 MB)
    char* ws = (char*)d_ws;
    const size_t XH_ELTS = (size_t)BDIM * HWDIM * CDIM;  // 4,194,304
    _Float16* xh    = (_Float16*)ws;                 ws += XH_ELTS * 2;
    _Float16* khat  = (_Float16*)ws;                 ws += XH_ELTS * 2;
    _Float16* qhat  = (_Float16*)ws;                 ws += XH_ELTS * 2;
    _Float16* kwh   = (_Float16*)ws;                 ws += (size_t)CDIM * CDIM * 2;
    _Float16* qwh   = (_Float16*)ws;                 ws += (size_t)CDIM * CDIM * 2;
    float*    vvals = (float*)ws;                    ws += (size_t)BDIM * HWDIM * 4;
    float*    wmap  = (float*)ws;

    // Stage 0: weights -> f16
    csa_cvt_w<<<dim3(CDIM * CDIM / 256), dim3(256), 0, stream>>>(key_w, kwh, CDIM * CDIM);
    csa_cvt_w<<<dim3(CDIM * CDIM / 256), dim3(256), 0, stream>>>(query_w, qwh, CDIM * CDIM);

    // Stage 1: transpose to [B,HW,C] f16
    csa_transpose<<<dim3(HWDIM / 32, CDIM / 32, BDIM), dim3(256), 0, stream>>>(batch, xh);

    // Stage 2: scalar value projection (f32)
    csa_values<<<dim3(HWDIM / 256, BDIM), dim3(256), 0, stream>>>(batch, value_w, value_b, vvals);

    // Stage 3: key/query projection + normalize (y: 0=key, 1=query)
    csa_proj_norm<<<dim3((BDIM * HWDIM) / 32, 2), dim3(256), 0, stream>>>(
        xh, kwh, qwh, key_b, query_b, khat, qhat);

    // Stage 4: column-softmax attention -> wmap [B,HW]
    csa_attention<<<dim3(HWDIM / 128, BDIM), dim3(256), 0, stream>>>(khat, qhat, vvals, wmap);

    // Stage 5: weighted 3x3 box sum -> output
    csa_finalize<<<dim3(CDIM, BDIM), dim3(256), 0, stream>>>(batch, wmap, out);
}
